// SDSCAM_8186207666338
// MI455X (gfx1250) — matrix operand order repaired and checked by reading, not compiled, not run on hardware
//
#include <hip/hip_runtime.h>
#include <math.h>

// ---------------------------------------------------------------------------
// SDSCAM for MI455X (gfx1250): bf16 WMMA everywhere, wave32, LDS-staged GEMMs.
// Shapes fixed by the reference: B=2, C=128, H=192, W=512.
// ---------------------------------------------------------------------------

#define Bn 2
#define Cc 128
#define Hh 192
#define Ww 512
#define HW (Hh * Ww)          // 98304
#define Pn (Bn * Hh * Ww)     // 196608 pixels

typedef __attribute__((ext_vector_type(16))) __bf16 v16bf;
typedef __attribute__((ext_vector_type(8)))  float  v8f;

// ---------------------------------------------------------------------------
// Fragment loaders per CDNA5 ISA 7.12.2 (wave32):
// A (16x32 MxK, bf16): lane l -> m = l&15, k0 = (l>>4)*8;
//   elements e=0..7 -> k=k0+e ; e=8..15 -> k=16+k0+(e-8)
// B (32x16 KxN, bf16) gathered from a [n][k] row-major source (weights stored
//   [out][in], K stored [key][c], Vt stored [c][key]):
//   lane l -> n = l&15, kb = (l>>4)*8; e=0..7 -> k=kb+e ; e=8..15 -> k=16+kb+(e-8)
// D/C (16x16 f32): element r -> m = r + 8*(l>>4), n = l&15
// ---------------------------------------------------------------------------
__device__ __forceinline__ v16bf frag_load_A(const __bf16* base, int ldm) {
  const int l = threadIdx.x & 31;
  const int m = l & 15, k0 = (l >> 4) * 8;
  const __bf16* r = base + (size_t)m * ldm;
  v16bf a;
#pragma unroll
  for (int e = 0; e < 8; ++e) a[e] = r[k0 + e];
#pragma unroll
  for (int e = 0; e < 8; ++e) a[8 + e] = r[16 + k0 + e];
  return a;
}

__device__ __forceinline__ v16bf frag_load_BT(const __bf16* base, int ldn) {
  const int l = threadIdx.x & 31;
  const int n = l & 15, kb = (l >> 4) * 8;
  const __bf16* r = base + (size_t)n * ldn + kb;
  v16bf b;
#pragma unroll
  for (int e = 0; e < 8; ++e) { b[e] = r[e]; b[8 + e] = r[16 + e]; }
  return b;
}

__device__ __forceinline__ v8f wmma_bf16(v16bf a, v16bf b, v8f c) {
  return __builtin_amdgcn_wmma_f32_16x16x32_bf16(false, a, false, b, (short)0, c,
                                                 false, false);
}

// ---------------------------------------------------------------------------
// 1) Channel LayerNorm: x [B,C,H,W] f32 -> N [B,H,W,C] bf16 (pixel-major).
//    128 threads/block, thread-per-pixel (channel reads coalesced across w),
//    LDS transpose so the bf16 stores are fully coalesced.
// ---------------------------------------------------------------------------
__global__ void ln_kernel(const float* __restrict__ x, const float* __restrict__ gw,
                          const float* __restrict__ gb, __bf16* __restrict__ nOut) {
  __shared__ __bf16 tile[128][Cc + 8];
  const int p0 = blockIdx.x * 128;
  const int p  = p0 + threadIdx.x;
  const int bb = p / HW;
  const int rem = p - bb * HW;
  const float* xp = x + (size_t)bb * Cc * HW + rem;

  float s = 0.f, ss = 0.f;
#pragma unroll 4
  for (int c = 0; c < Cc; ++c) {
    float v = xp[(size_t)c * HW];
    s += v; ss += v * v;
  }
  const float mu  = s * (1.f / Cc);
  const float var = ss * (1.f / Cc) - mu * mu;
  const float inv = rsqrtf(var + 1e-6f);

#pragma unroll 4
  for (int c = 0; c < Cc; ++c) {
    float v = (xp[(size_t)c * HW] - mu) * inv * gw[c] + gb[c];
    tile[threadIdx.x][c] = (__bf16)v;
  }
  __syncthreads();
  __bf16* op = nOut + (size_t)p0 * Cc;
#pragma unroll 4
  for (int i = 0; i < 128; ++i)  // row i = pixel i of tile, col = channel
    op[(size_t)i * Cc + threadIdx.x] = tile[i][threadIdx.x];
}

// ---------------------------------------------------------------------------
// 2) Weight repack to bf16 (1x1: [O][C] kept; dconv: [O][C][3][3] -> [9][O][C])
// ---------------------------------------------------------------------------
__global__ void pack_w_kernel(const float* __restrict__ w, __bf16* __restrict__ o, int n) {
  int i = blockIdx.x * 256 + threadIdx.x;
  if (i < n) o[i] = (__bf16)w[i];
}

__global__ void pack_dw_kernel(const float* __restrict__ w, __bf16* __restrict__ o) {
  int i = blockIdx.x * 256 + threadIdx.x;   // over O*C*9
  if (i >= Cc * Cc * 9) return;
  int t = i % 9, oc = i / 9;                // w layout: ((o*C+c)*3+kh)*3+kw
  o[(size_t)t * Cc * Cc + oc] = (__bf16)w[i];
}

// ---------------------------------------------------------------------------
// 3) 1x1 conv as GEMM: Out[p,o] = sum_c In[p,c] * Wt[o,c] + bias[o]
//    256 thr = 8 waves; tile 64 px x 128 out; wave -> 16-out column, 4 m-tiles.
//    SRC_BF16: In is N bf16 [P,C]; else In is original f32 x [B,C,H,W].
// ---------------------------------------------------------------------------
template <bool SRC_BF16>
__global__ void conv1x1_kernel(const void* __restrict__ src,
                               const __bf16* __restrict__ wt,
                               const float* __restrict__ bias,
                               __bf16* __restrict__ out) {
  __shared__ __bf16 As[64][Cc + 8];
  __shared__ __bf16 Bs[Cc][Cc + 8];
  const int p0  = blockIdx.x * 64;
  const int tid = threadIdx.x;

  for (int i = tid; i < Cc * Cc; i += 256) Bs[i >> 7][i & 127] = wt[i];

  if (SRC_BF16) {
    const __bf16* s = (const __bf16*)src + (size_t)p0 * Cc;
    for (int i = tid; i < 64 * Cc; i += 256) As[i >> 7][i & 127] = s[i];
  } else {
    const float* xs = (const float*)src;
    const int bb = p0 / HW, rem = p0 - bb * HW;
    const float* base = xs + (size_t)bb * Cc * HW + rem;
    for (int i = tid; i < 64 * Cc; i += 256) {
      int c = i >> 6, px = i & 63;
      As[px][c] = (__bf16)base[(size_t)c * HW + px];
    }
  }
  __syncthreads();

  const int wv = tid >> 5, lane = tid & 31;
  const int nblk = wv * 16;
  v16bf bfr[4];
#pragma unroll
  for (int ks = 0; ks < 4; ++ks) bfr[ks] = frag_load_BT(&Bs[nblk][ks * 32], Cc + 8);

  const int n = lane & 15, hi = lane >> 4;
  const float bv = bias[nblk + n];
#pragma unroll
  for (int mt = 0; mt < 4; ++mt) {
    v8f acc = {0.f, 0.f, 0.f, 0.f, 0.f, 0.f, 0.f, 0.f};
#pragma unroll
    for (int ks = 0; ks < 4; ++ks)
      acc = wmma_bf16(frag_load_A(&As[mt * 16][ks * 32], Cc + 8), bfr[ks], acc);
    __bf16* op = out + (size_t)(p0 + mt * 16) * Cc + nblk + n;
#pragma unroll
    for (int r = 0; r < 8; ++r)
      op[(size_t)(r + 8 * hi) * Cc] = (__bf16)(acc[r] + bv);
  }
}

// ---------------------------------------------------------------------------
// 4) Dilated 3x3 conv (dilation 4, pad 4) = 9 shifted GEMM taps on N [P,C].
//    wt9: [9][O][C] bf16. Same tiling as conv1x1; zero-fill OOB taps.
// ---------------------------------------------------------------------------
__global__ void dconv_kernel(const __bf16* __restrict__ nsrc,
                             const __bf16* __restrict__ wt9,
                             const float* __restrict__ bias,
                             __bf16* __restrict__ out) {
  __shared__ __bf16 As[64][Cc + 8];
  __shared__ __bf16 Bs[Cc][Cc + 8];
  const int p0  = blockIdx.x * 64;
  const int bb  = p0 / HW;
  const int rem = p0 - bb * HW;
  const int h   = rem / Ww;
  const int w0  = rem - h * Ww;
  const int tid = threadIdx.x;
  const int wv = tid >> 5, lane = tid & 31, nblk = wv * 16;

  v8f acc[4];
#pragma unroll
  for (int mt = 0; mt < 4; ++mt)
#pragma unroll
    for (int j = 0; j < 8; ++j) acc[mt][j] = 0.f;

  for (int tap = 0; tap < 9; ++tap) {
    const int dh = (tap / 3 - 1) * 4, dw = (tap % 3 - 1) * 4;
    const int hh = h + dh;
    const bool rowok = (hh >= 0) && (hh < Hh);
    __syncthreads();  // previous tap's fragment reads complete before restage

    const __bf16* wp = wt9 + (size_t)tap * Cc * Cc;
    if (tap < 8) __builtin_prefetch(wp + Cc * Cc, 0, 0);  // global_prefetch_b8
    for (int i = tid; i < Cc * Cc; i += 256) Bs[i >> 7][i & 127] = wp[i];

    const __bf16 z = (__bf16)0.0f;
    for (int i = tid; i < 64 * Cc; i += 256) {
      int px = i >> 7, c = i & 127;
      int ww2 = w0 + px + dw;
      __bf16 v = z;
      if (rowok && ww2 >= 0 && ww2 < Ww)
        v = nsrc[((size_t)(bb * Hh + hh) * Ww + ww2) * Cc + c];
      As[px][c] = v;
    }
    __syncthreads();

    v16bf bfr[4];
#pragma unroll
    for (int ks = 0; ks < 4; ++ks) bfr[ks] = frag_load_BT(&Bs[nblk][ks * 32], Cc + 8);
#pragma unroll
    for (int mt = 0; mt < 4; ++mt)
#pragma unroll
      for (int ks = 0; ks < 4; ++ks)
        acc[mt] = wmma_bf16(frag_load_A(&As[mt * 16][ks * 32], Cc + 8), bfr[ks], acc[mt]);
  }

  const int n = lane & 15, hi = lane >> 4;
  const float bv = bias[nblk + n];
#pragma unroll
  for (int mt = 0; mt < 4; ++mt) {
    __bf16* op = out + (size_t)(p0 + mt * 16) * Cc + nblk + n;
#pragma unroll
    for (int r = 0; r < 8; ++r)
      op[(size_t)(r + 8 * hi) * Cc] = (__bf16)(acc[mt][r] + bv);
  }
}

// ---------------------------------------------------------------------------
// 5) Per-row flash attention + residual epilogue.
//    Block = 128 thr = 4 waves; grid = B*H*(W/64). Each wave owns 16 query
//    rows; loop over 8 key chunks of 64 with online softmax (half-wave
//    shfl_xor row reductions). Epilogue: out[b,c,h,w] = x + O*beta (coalesced
//    via LDS transpose, overlaying the K/V staging buffers).
// ---------------------------------------------------------------------------
#define NK 64
__global__ void attn_kernel(const __bf16* __restrict__ Q, const __bf16* __restrict__ K,
                            const __bf16* __restrict__ V, const float* __restrict__ x,
                            const float* __restrict__ betac, float* __restrict__ out) {
  // manual LDS layout: [Ks 64x136 bf16 | Vt 128x72 bf16] overlaid by Os 64x129 f32
  __shared__ __align__(16) char smem[64 * 136 * 2 + 128 * 72 * 2];
  __shared__ __bf16 Ps[4][16][NK + 8];
  __bf16 (*Ks)[Cc + 8] = (__bf16(*)[Cc + 8])smem;                 // [key][c]
  __bf16 (*Vt)[NK + 8] = (__bf16(*)[NK + 8])(smem + 64 * 136 * 2); // [c][key]
  float (*Os)[Cc + 1]  = (float(*)[Cc + 1])smem;                   // epilogue overlay

  const int blk  = blockIdx.x;
  const int wt   = blk & 7;         // W/64 tiles
  const int bh   = blk >> 3;        // b*H + h
  const int w0   = wt * 64;
  const size_t rowOff = (size_t)bh * Ww * Cc;
  const int tid  = threadIdx.x;
  const int wv   = tid >> 5, lane = tid & 31;
  const int n    = lane & 15, hi = lane >> 4;
  const float scale = 0.08838834764831845f;  // 128^-0.5

  // Q A-fragments for this wave's 16 rows (held in registers for all chunks)
  v16bf qf[4];
  const __bf16* qbase = Q + rowOff + (size_t)(w0 + wv * 16) * Cc;
#pragma unroll
  for (int ks = 0; ks < 4; ++ks) qf[ks] = frag_load_A(qbase + ks * 32, Cc);

  float rmax[8], rsum[8];
  v8f o[8];
#pragma unroll
  for (int r = 0; r < 8; ++r) { rmax[r] = -3.0e38f; rsum[r] = 0.f; }
#pragma unroll
  for (int cf = 0; cf < 8; ++cf)
#pragma unroll
    for (int r = 0; r < 8; ++r) o[cf][r] = 0.f;

  for (int ch = 0; ch < Ww / NK; ++ch) {
    const int k0c = ch * NK;
    __syncthreads();  // all waves done reading previous chunk's Ks/Vt
    for (int i = tid; i < NK * Cc; i += 128) {
      int key = i >> 7, c = i & 127;
      Ks[key][c] = K[rowOff + (size_t)(k0c + key) * Cc + c];
    }
    for (int i = tid; i < NK * Cc; i += 128) {
      int key = i >> 7, c = i & 127;
      Vt[c][key] = V[rowOff + (size_t)(k0c + key) * Cc + c];
    }
    __syncthreads();

    // S = scale * Q K^T for this wave's 16 rows x 64 keys
    v8f s[4];
#pragma unroll
    for (int nf = 0; nf < 4; ++nf) {
      v8f a = {0.f, 0.f, 0.f, 0.f, 0.f, 0.f, 0.f, 0.f};
#pragma unroll
      for (int ks = 0; ks < 4; ++ks)
        a = wmma_bf16(qf[ks], frag_load_BT(&Ks[nf * 16][ks * 32], Cc + 8), a);
#pragma unroll
      for (int r = 0; r < 8; ++r) a[r] *= scale;
      s[nf] = a;
    }

    // per-row chunk max (half-wave holds 16 distinct key columns per row)
    float cmax[8];
#pragma unroll
    for (int r = 0; r < 8; ++r) {
      float m0 = s[0][r];
#pragma unroll
      for (int nf = 1; nf < 4; ++nf) m0 = fmaxf(m0, s[nf][r]);
      cmax[r] = m0;
    }
#pragma unroll
    for (int off = 1; off < 16; off <<= 1)
#pragma unroll
      for (int r = 0; r < 8; ++r)
        cmax[r] = fmaxf(cmax[r], __shfl_xor(cmax[r], off, 32));

    float ps[8];
#pragma unroll
    for (int r = 0; r < 8; ++r) {
      const float nm   = fmaxf(rmax[r], cmax[r]);
      const float corr = __expf(rmax[r] - nm);
      rmax[r] = nm;
      rsum[r] *= corr;
#pragma unroll
      for (int cf = 0; cf < 8; ++cf) o[cf][r] *= corr;
      float acc = 0.f;
#pragma unroll
      for (int nf = 0; nf < 4; ++nf) {
        float p = __expf(s[nf][r] - nm);
        acc += p;
        s[nf][r] = p;
      }
      ps[r] = acc;
    }
#pragma unroll
    for (int off = 1; off < 16; off <<= 1)
#pragma unroll
      for (int r = 0; r < 8; ++r) ps[r] += __shfl_xor(ps[r], off, 32);
#pragma unroll
    for (int r = 0; r < 8; ++r) rsum[r] += ps[r];

    // P (D-layout) -> LDS -> A-layout fragments; same-wave DS ordering
#pragma unroll
    for (int nf = 0; nf < 4; ++nf)
#pragma unroll
      for (int r = 0; r < 8; ++r)
        Ps[wv][r + 8 * hi][nf * 16 + n] = (__bf16)s[nf][r];

    v16bf pf[2];
#pragma unroll
    for (int ks = 0; ks < 2; ++ks) pf[ks] = frag_load_A(&Ps[wv][0][ks * 32], NK + 8);
#pragma unroll
    for (int cf = 0; cf < 8; ++cf)
#pragma unroll
      for (int ks = 0; ks < 2; ++ks)
        o[cf] = wmma_bf16(pf[ks], frag_load_BT(&Vt[cf * 16][ks * 32], NK + 8), o[cf]);
  }

  __syncthreads();  // K/V staging dead -> overlay with Os
  float rinv[8];
#pragma unroll
  for (int r = 0; r < 8; ++r) rinv[r] = 1.f / rsum[r];
#pragma unroll
  for (int cf = 0; cf < 8; ++cf)
#pragma unroll
    for (int r = 0; r < 8; ++r)
      Os[wv * 16 + r + 8 * hi][cf * 16 + n] = o[cf][r] * rinv[r];
  __syncthreads();

  const int b = bh / Hh, h = bh - b * Hh;
  for (int i = tid; i < 64 * Cc; i += 128) {
    int c = i >> 6, px = i & 63;
    size_t g = (((size_t)b * Cc + c) * Hh + h) * Ww + w0 + px;
    out[g] = x[g] + Os[px][c] * betac[c];
  }
}

// ---------------------------------------------------------------------------
// Host-side launch
// ---------------------------------------------------------------------------
extern "C" void kernel_launch(void* const* d_in, const int* in_sizes, int n_in,
                              void* d_out, int out_size, void* d_ws, size_t ws_size,
                              hipStream_t stream) {
  const float* x_l  = (const float*)d_in[0];
  const float* x_r  = (const float*)d_in[1];
  const float* nl_w = (const float*)d_in[2];
  const float* nl_b = (const float*)d_in[3];
  const float* nr_w = (const float*)d_in[4];
  const float* nr_b = (const float*)d_in[5];
  const float* lp1w = (const float*)d_in[6];
  const float* lp1b = (const float*)d_in[7];
  const float* rp1w = (const float*)d_in[8];
  const float* rp1b = (const float*)d_in[9];
  const float* dlw  = (const float*)d_in[10];
  const float* dlb  = (const float*)d_in[11];
  const float* drw  = (const float*)d_in[12];
  const float* drb  = (const float*)d_in[13];
  const float* lp2w = (const float*)d_in[14];
  const float* lp2b = (const float*)d_in[15];
  const float* rp2w = (const float*)d_in[16];
  const float* rp2b = (const float*)d_in[17];
  const float* beta = (const float*)d_in[18];
  const float* gamm = (const float*)d_in[19];

  // workspace layout (bf16 tensors are [P, C] pixel-major)
  const size_t SZ = (size_t)Pn * Cc * sizeof(__bf16);   // 50,331,648 B
  char* ws = (char*)d_ws;
  __bf16* N_l = (__bf16*)(ws + 0 * SZ);
  __bf16* N_r = (__bf16*)(ws + 1 * SZ);
  __bf16* Q_l = (__bf16*)(ws + 2 * SZ);
  __bf16* Q_r = (__bf16*)(ws + 3 * SZ);
  __bf16* V_l = (__bf16*)(ws + 4 * SZ);
  __bf16* V_r = (__bf16*)(ws + 5 * SZ);
  __bf16* K_l = (__bf16*)(ws + 6 * SZ);
  __bf16* K_r = (__bf16*)(ws + 7 * SZ);
  char* wsp = ws + 8 * SZ;
  const size_t W1 = (size_t)Cc * Cc * sizeof(__bf16);   // 32 KB
  __bf16* lp1bf = (__bf16*)(wsp);            wsp += W1;
  __bf16* rp1bf = (__bf16*)(wsp);            wsp += W1;
  __bf16* lp2bf = (__bf16*)(wsp);            wsp += W1;
  __bf16* rp2bf = (__bf16*)(wsp);            wsp += W1;
  __bf16* dlbf  = (__bf16*)(wsp);            wsp += 9 * W1;
  __bf16* drbf  = (__bf16*)(wsp);

  float* out0 = (float*)d_out;
  float* out1 = out0 + (size_t)Bn * Cc * HW;

  // 1) weight repack
  pack_w_kernel<<<64, 256, 0, stream>>>(lp1w, lp1bf, Cc * Cc);
  pack_w_kernel<<<64, 256, 0, stream>>>(rp1w, rp1bf, Cc * Cc);
  pack_w_kernel<<<64, 256, 0, stream>>>(lp2w, lp2bf, Cc * Cc);
  pack_w_kernel<<<64, 256, 0, stream>>>(rp2w, rp2bf, Cc * Cc);
  pack_dw_kernel<<<(Cc * Cc * 9 + 255) / 256, 256, 0, stream>>>(dlw, dlbf);
  pack_dw_kernel<<<(Cc * Cc * 9 + 255) / 256, 256, 0, stream>>>(drw, drbf);

  // 2) LayerNorm
  ln_kernel<<<Pn / 128, 128, 0, stream>>>(x_l, nl_w, nl_b, N_l);
  ln_kernel<<<Pn / 128, 128, 0, stream>>>(x_r, nr_w, nr_b, N_r);

  // 3) 1x1 convs (Q from normalized bf16; V from raw fp32 x)
  conv1x1_kernel<true ><<<Pn / 64, 256, 0, stream>>>(N_l, lp1bf, lp1b, Q_l);
  conv1x1_kernel<true ><<<Pn / 64, 256, 0, stream>>>(N_r, rp1bf, rp1b, Q_r);
  conv1x1_kernel<false><<<Pn / 64, 256, 0, stream>>>(x_l, lp2bf, lp2b, V_l);
  conv1x1_kernel<false><<<Pn / 64, 256, 0, stream>>>(x_r, rp2bf, rp2b, V_r);

  // 4) dilated 3x3 convs -> K
  dconv_kernel<<<Pn / 64, 256, 0, stream>>>(N_l, dlbf, dlb, K_l);
  dconv_kernel<<<Pn / 64, 256, 0, stream>>>(N_r, drbf, drb, K_r);

  // 5) cross attention + fused residual
  const int agrid = Bn * Hh * (Ww / 64);
  attn_kernel<<<agrid, 128, 0, stream>>>(Q_l, K_r, V_r, x_l, beta, out0);
  attn_kernel<<<agrid, 128, 0, stream>>>(Q_r, K_l, V_l, x_r, gamm, out1);
}
